// Siege_25116968747557
// MI455X (gfx1250) — matrix operand order repaired and checked by reading, not compiled, not run on hardware
//
#include <hip/hip_runtime.h>
#include <hip/hip_bf16.h>

// ---------------------------------------------------------------------------
// CDNA5 (gfx1250) implementation of the equivariant GNN denoiser forward pass.
// All dense GEMMs are routed through V_WMMA_F32_16X16X32_BF16 (wave32).
//
// GEMM strategy: weights are pre-transposed to bf16 B^T (N x K), so BOTH
// fragments are contiguous 16B runs loaded directly from global memory
// (L2-resident weights) -- no LDS, no barriers, no staging.  Optional
// row-index indirection on A (fused graph gather) and fused per-(row,col)
// scale on the epilogue (fused edge-weight multiply).
// ---------------------------------------------------------------------------

typedef __bf16 bf16_t;
typedef __attribute__((ext_vector_type(16))) __bf16 v16bf;
typedef __attribute__((ext_vector_type(8)))  __bf16 v8bf;
typedef __attribute__((ext_vector_type(8)))  float  v8f;

#define ACT_NONE 0
#define ACT_SILU 1
#define ACT_SIG  2

__device__ __forceinline__ bf16_t f2bf(float f) {
  union { float f; unsigned int u; } x; x.f = f;
  unsigned int r = (x.u + 0x7FFFu + ((x.u >> 16) & 1u)) >> 16;
  union { unsigned short s; bf16_t b; } y; y.s = (unsigned short)r;
  return y.b;
}
__device__ __forceinline__ float bf2f(bf16_t b) {
  union { bf16_t b; unsigned short s; } x; x.b = b;
  union { unsigned int u; float f; } y; y.u = ((unsigned int)x.s) << 16;
  return y.f;
}
__device__ __forceinline__ float siluf(float x) { return x / (1.0f + __expf(-x)); }
__device__ __forceinline__ float sigm(float x)  { return 1.0f / (1.0f + __expf(-x)); }

__device__ __forceinline__ void atomicMaxF(float* addr, float val) {
  unsigned int* ua = (unsigned int*)addr;
  unsigned int old = *ua;
  while (true) {
    float f = __uint_as_float(old);
    if (f >= val) break;
    unsigned int assumed = old;
    old = atomicCAS(ua, assumed, __float_as_uint(val));
    if (old == assumed) break;
  }
}

// ---------------------------------------------------------------------------
// WMMA GEMM: C[M,N] = act(A[Aidx[M],K] @ B[K,N]) (*emul) (+C)
//   A    : row-major bf16 (rows x K),  K % 32 == 0
//   Aidx : optional row indices (fused gather); nullptr = identity
//   BT   : B TRANSPOSED, row-major bf16 (N x K)  [weights, L2-resident]
//   emul : optional f32 scale, val *= emul[row*emul_ld + emul_off + col]
//          (applied AFTER the optional accumulate-add: (Cold+val)*emul)
//
// Block = 256 threads = 8 waves; wave w computes a 16(M) x 64(N) strip
// (4 accumulators): the A-fragment is reused across 4 WMMAs per K-step.
// Grid: x = ceil(M/128), y = ceil(N/64).
//
// A-fragment (ISA 7.12.2, 16-bit A 16x32): lane l holds row (l&15), K elems
// {koff..koff+7, 16+koff..+7}, koff = 8*(l>>4) -> two contiguous b128 loads.
// B-fragment (16-bit B 32x16): lane l holds col (l&15), K rows
// {koff..koff+7, 16+koff..+7} (same as A) -> two b128 loads from B^T.
// ---------------------------------------------------------------------------
__global__ __launch_bounds__(256)
void sg_wmma_gemm(const bf16_t* __restrict__ A, const int* __restrict__ Aidx,
                  const bf16_t* __restrict__ BT, void* __restrict__ Cout,
                  int M, int N, int K, int act, int accumulate, int out_bf16,
                  const float* __restrict__ emul, int emul_ld, int emul_off) {
  const int lane = threadIdx.x & 31;
  const int wave = threadIdx.x >> 5;
  const int m0 = blockIdx.x * 128 + wave * 16;
  const int n0 = blockIdx.y * 64;

  const int am    = m0 + (lane & 15);
  const int koff  = (lane >> 4) * 8;
  const int rbase = (lane >> 4) * 8;
  const int bn    = lane & 15;

  v8f acc[4] = {};

  const bool interior = (m0 + 16 <= M) && (n0 + 64 <= N);
  const bool aok = (am < M);
  const int arowi = Aidx ? (aok ? Aidx[am] : 0) : am;
  const bf16_t* __restrict__ arow = A + (size_t)arowi * K;

  if (interior) {
    for (int k0 = 0; k0 < K; k0 += 32) {
      union { v16bf v; v8bf h[2]; } af;
      af.h[0] = *(const v8bf*)(arow + k0 + koff);
      af.h[1] = *(const v8bf*)(arow + k0 + 16 + koff);
      if (k0 + 32 < K) __builtin_prefetch(arow + k0 + 32 + koff, 0, 1);
#pragma unroll
      for (int t4 = 0; t4 < 4; ++t4) {
        const bf16_t* br = BT + (size_t)(n0 + t4 * 16 + bn) * K + k0 + rbase;
        union { v16bf v; v8bf h[2]; } bfr;
        bfr.h[0] = *(const v8bf*)(br);
        bfr.h[1] = *(const v8bf*)(br + 16);
        acc[t4] = __builtin_amdgcn_wmma_f32_16x16x32_bf16(
            false, af.v, false, bfr.v, (short)0, acc[t4], false, false);
      }
    }
  } else {
    // guarded tail path: OOB lanes get zeros (their outputs are never stored)
    for (int k0 = 0; k0 < K; k0 += 32) {
      union { v16bf v; v8bf h[2]; } af;
      if (aok) {
        af.h[0] = *(const v8bf*)(arow + k0 + koff);
        af.h[1] = *(const v8bf*)(arow + k0 + 16 + koff);
      } else {
#pragma unroll
        for (int j = 0; j < 16; ++j) af.v[j] = f2bf(0.0f);
      }
#pragma unroll
      for (int t4 = 0; t4 < 4; ++t4) {
        const int col = n0 + t4 * 16 + bn;
        union { v16bf v; v8bf h[2]; } bfr;
        if (col < N) {
          const bf16_t* br = BT + (size_t)col * K + k0 + rbase;
          bfr.h[0] = *(const v8bf*)(br);
          bfr.h[1] = *(const v8bf*)(br + 16);
        } else {
#pragma unroll
          for (int j = 0; j < 16; ++j) bfr.v[j] = f2bf(0.0f);
        }
        acc[t4] = __builtin_amdgcn_wmma_f32_16x16x32_bf16(
            false, af.v, false, bfr.v, (short)0, acc[t4], false, false);
      }
    }
  }

  // D layout: VGPR r -> row r + 8*(lane>>4), col lane&15
#pragma unroll
  for (int t4 = 0; t4 < 4; ++t4) {
    const int cn = n0 + t4 * 16 + (lane & 15);
    if (cn >= N) continue;
#pragma unroll
    for (int r = 0; r < 8; ++r) {
      const int cm = m0 + r + 8 * (lane >> 4);
      if (cm < M) {
        float val = acc[t4][r];
        if (act == ACT_SILU) val = siluf(val);
        else if (act == ACT_SIG) val = sigm(val);
        const size_t idx = (size_t)cm * N + cn;
        if (out_bf16) {
          ((bf16_t*)Cout)[idx] = f2bf(val);
        } else {
          float* Cf = (float*)Cout;
          if (accumulate) val += Cf[idx];
          if (emul) val *= emul[(size_t)cm * emul_ld + emul_off + cn];
          Cf[idx] = val;
        }
      }
    }
  }
}

// ------------------------------ utility kernels ----------------------------
__global__ void sg_f32_to_bf16(const float* __restrict__ src,
                               bf16_t* __restrict__ dst, size_t n) {
  size_t i = (size_t)blockIdx.x * blockDim.x + threadIdx.x;
  if (i < n) dst[i] = f2bf(src[i]);
}

// batched weight convert + transpose: src nmat x (K x N) -> dst nmat x (N x K)
__global__ void sg_cvt_w_t(const float* __restrict__ src, bf16_t* __restrict__ dst,
                           int K, int N, int nmat) {
  size_t i = (size_t)blockIdx.x * blockDim.x + threadIdx.x;
  size_t per = (size_t)K * N;
  if (i >= per * (size_t)nmat) return;
  size_t mat = i / per; size_t r = i % per;
  int k = (int)(r / N), n = (int)(r % N);
  dst[mat * per + (size_t)n * K + k] = f2bf(src[i]);
}

__global__ void sg_fill(float* __restrict__ p, float v, size_t n) {
  size_t i = (size_t)blockIdx.x * blockDim.x + threadIdx.x;
  if (i < n) p[i] = v;
}

// ------------------------------ model kernels ------------------------------
__global__ void sg_node_embed(const float* __restrict__ t,
                              const int* __restrict__ f_in,
                              const float* __restrict__ W,  // 84x128
                              float* __restrict__ s, int n) {
  __shared__ float temb[64];
  int nid = blockIdx.x; if (nid >= n) return;
  int tid = threadIdx.x;
  if (tid < 32) {
    float fr = __expf(-logf(10000.0f) * (float)tid / 32.0f);
    float ang = t[nid] * 10000.0f * fr;
    temb[tid]      = __sinf(ang);
    temb[tid + 32] = __cosf(ang);
  }
  __syncthreads();
  float acc = W[(size_t)f_in[nid] * 128 + tid];
  for (int k = 0; k < 64; ++k) acc += temb[k] * W[(size_t)(20 + k) * 128 + tid];
  s[(size_t)nid * 128 + tid] = acc;
}

__global__ void sg_edge_geom(const float* __restrict__ pos,
                             const int* __restrict__ src,
                             const int* __restrict__ dst,
                             float* __restrict__ sh,    // E x 8 : [sh1(3), sh2(5)]
                             bf16_t* __restrict__ rbf,  // E x 128
                             int E) {
  int e = blockIdx.x; if (e >= E) return;
  int tid = threadIdx.x;
  int a = src[e], b = dst[e];
  float vx = pos[a * 3 + 0] - pos[b * 3 + 0];
  float vy = pos[a * 3 + 1] - pos[b * 3 + 1];
  float vz = pos[a * 3 + 2] - pos[b * 3 + 2];
  float d = sqrtf(vx * vx + vy * vy + vz * vz);
  if (tid == 0) {
    float inv = 1.0f / (d + 1e-9f);
    float x = vx * inv, y = vy * inv, z = vz * inv;
    const float s3 = 1.7320508f, c15 = 3.8729833f, c5 = 2.2360680f;
    sh[(size_t)e * 8 + 0] = s3 * y;
    sh[(size_t)e * 8 + 1] = s3 * z;
    sh[(size_t)e * 8 + 2] = s3 * x;
    sh[(size_t)e * 8 + 3] = c15 * x * y;
    sh[(size_t)e * 8 + 4] = c15 * y * z;
    sh[(size_t)e * 8 + 5] = 0.5f * c5 * (3.0f * z * z - 1.0f);
    sh[(size_t)e * 8 + 6] = c15 * x * z;
    sh[(size_t)e * 8 + 7] = 0.5f * c15 * (x * x - y * y);
  }
  float center = 30.0f * (float)tid / 127.0f;
  float step = 30.0f / 128.0f;
  float gamma = 0.5f / (step * step);
  float diff = d - center;
  rbf[(size_t)e * 128 + tid] = f2bf(__expf(-gamma * diff * diff));
}

__global__ void sg_scatter_init(const float* __restrict__ w,   // E x 224
                                const float* __restrict__ sh,
                                const int* __restrict__ dst,
                                float* __restrict__ s, float* __restrict__ v,
                                float* __restrict__ q, int E) {
  int e = blockIdx.x; if (e >= E) return;
  int j = threadIdx.x; if (j >= 480) return;
  const float inv = 0.2533530f;  // 1/sqrt(15.5793085)
  int d = dst[e];
  if (j < 128) {
    atomicAdd(&s[(size_t)d * 128 + j], w[(size_t)e * 224 + j] * inv);
  } else if (j < 320) {
    int u = j - 128; int m = u / 3, c = u % 3;
    atomicAdd(&v[(size_t)d * 192 + u],
              w[(size_t)e * 224 + 128 + m] * sh[(size_t)e * 8 + c] * inv);
  } else {
    int u = j - 320; int m = u / 5, c = u % 5;
    atomicAdd(&q[(size_t)d * 160 + u],
              w[(size_t)e * 224 + 192 + m] * sh[(size_t)e * 8 + 3 + c] * inv);
  }
}

__global__ __launch_bounds__(128)
void sg_eqln(const float* __restrict__ s, const float* __restrict__ v,
             const float* __restrict__ q, const float* __restrict__ gs,
             const float* __restrict__ gv, const float* __restrict__ gq,
             float* __restrict__ ys, bf16_t* __restrict__ ysb,
             float* __restrict__ yv, float* __restrict__ yq, int n) {
  __shared__ float red[128];
  int nid = blockIdx.x; if (nid >= n) return;
  int t = threadIdx.x;
  // scalar channel
  float sv_ = s[(size_t)nid * 128 + t];
  red[t] = sv_; __syncthreads();
  for (int st = 64; st > 0; st >>= 1) { if (t < st) red[t] += red[t + st]; __syncthreads(); }
  float mean = red[0] / 128.0f; __syncthreads();
  float s0 = sv_ - mean;
  red[t] = s0 * s0; __syncthreads();
  for (int st = 64; st > 0; st >>= 1) { if (t < st) red[t] += red[t + st]; __syncthreads(); }
  float rms = red[0] / 128.0f; __syncthreads();
  float o = s0 * rsqrtf(rms + 1e-5f) * gs[t];
  ys[(size_t)nid * 128 + t] = o;
  ysb[(size_t)nid * 128 + t] = f2bf(o);
  // vector channel (64 x 3)
  float nv = 0.0f;
  if (t < 64) {
    for (int c = 0; c < 3; ++c) { float x = v[(size_t)nid * 192 + t * 3 + c]; nv += x * x; }
  }
  red[t] = (t < 64) ? nv : 0.0f; __syncthreads();
  for (int st = 64; st > 0; st >>= 1) { if (t < st) red[t] += red[t + st]; __syncthreads(); }
  float vn = red[0] / 64.0f; __syncthreads();
  if (t < 64) {
    float sc = rsqrtf(vn + 1e-5f) * gv[t];
    for (int c = 0; c < 3; ++c)
      yv[(size_t)nid * 192 + t * 3 + c] = v[(size_t)nid * 192 + t * 3 + c] * sc;
  }
  // quadrupole channel (32 x 5)
  float nq = 0.0f;
  if (t < 32) {
    for (int c = 0; c < 5; ++c) { float x = q[(size_t)nid * 160 + t * 5 + c]; nq += x * x; }
  }
  red[t] = (t < 32) ? nq : 0.0f; __syncthreads();
  for (int st = 64; st > 0; st >>= 1) { if (t < st) red[t] += red[t + st]; __syncthreads(); }
  float qn = red[0] / 32.0f; __syncthreads();
  if (t < 32) {
    float sc = rsqrtf(qn + 1e-5f) * gq[t];
    for (int c = 0; c < 5; ++c)
      yq[(size_t)nid * 160 + t * 5 + c] = q[(size_t)nid * 160 + t * 5 + c] * sc;
  }
}

__global__ void sg_mv_edge(const float* __restrict__ sv, const float* __restrict__ wr,
                           const float* __restrict__ sh, const float* __restrict__ yv,
                           const int* __restrict__ src, const float* __restrict__ Wv,
                           float* __restrict__ mv, int E) {
  size_t i = (size_t)blockIdx.x * blockDim.x + threadIdx.x;
  if (i >= (size_t)E * 192) return;
  size_t e = i / 192; int u = (int)(i % 192); int k = u / 3, c = u % 3;
  float a = sv[e * 64 + k] * wr[e * 320 + 128 + k] * sh[e * 8 + c];
  const float* yvp = yv + (size_t)src[e] * 192 + c;
  float acc = 0.0f;
#pragma unroll 4
  for (int m = 0; m < 64; ++m) acc += yvp[m * 3] * Wv[m * 64 + k];
  mv[i] = a + acc * wr[e * 320 + 192 + k];
}

__global__ void sg_mq_edge(const float* __restrict__ sq, const float* __restrict__ wr,
                           const float* __restrict__ sh, const float* __restrict__ yq,
                           const int* __restrict__ src, const float* __restrict__ Wq,
                           float* __restrict__ mq, int E) {
  size_t i = (size_t)blockIdx.x * blockDim.x + threadIdx.x;
  if (i >= (size_t)E * 160) return;
  size_t e = i / 160; int u = (int)(i % 160); int k = u / 5, c = u % 5;
  float a = sq[e * 32 + k] * wr[e * 320 + 256 + k] * sh[e * 8 + 3 + c];
  const float* yqp = yq + (size_t)src[e] * 160 + c;
  float acc = 0.0f;
#pragma unroll 4
  for (int m = 0; m < 32; ++m) acc += yqp[m * 5] * Wq[m * 32 + k];
  mq[i] = a + acc * wr[e * 320 + 288 + k];
}

__global__ void sg_logits(const float* __restrict__ ms, const float* __restrict__ A,
                          float* __restrict__ logits, int E) {
  size_t i = (size_t)blockIdx.x * blockDim.x + threadIdx.x;
  if (i >= (size_t)E * 4) return;
  size_t e = i / 4; int h = (int)(i % 4);
  float acc = 0.0f;
  for (int d = 0; d < 32; ++d) {
    float x = ms[e * 128 + h * 32 + d];
    x = (x > 0.0f) ? x : 0.2f * x;  // leaky_relu 0.2
    acc += x * A[h * 32 + d];
  }
  logits[i] = acc;
}

__global__ void sg_segmax(const float* __restrict__ logits, const int* __restrict__ dst,
                          float* __restrict__ nodemax, int E) {
  size_t i = (size_t)blockIdx.x * blockDim.x + threadIdx.x;
  if (i >= (size_t)E * 4) return;
  size_t e = i / 4; int h = (int)(i % 4);
  atomicMaxF(&nodemax[(size_t)dst[e] * 4 + h], logits[i]);
}

__global__ void sg_segexp(const float* __restrict__ logits, const int* __restrict__ dst,
                          const float* __restrict__ nodemax, float* __restrict__ nodesum,
                          float* __restrict__ attn, int E) {
  size_t i = (size_t)blockIdx.x * blockDim.x + threadIdx.x;
  if (i >= (size_t)E * 4) return;
  size_t e = i / 4; int h = (int)(i % 4);
  float ex = __expf(logits[i] - nodemax[(size_t)dst[e] * 4 + h]);
  attn[i] = ex;
  atomicAdd(&nodesum[(size_t)dst[e] * 4 + h], ex);
}

__global__ void sg_scatter_attn(const float* __restrict__ ms, const float* __restrict__ mv,
                                const float* __restrict__ mq, const float* __restrict__ attn,
                                const float* __restrict__ nodesum, const int* __restrict__ dst,
                                float* __restrict__ as_, float* __restrict__ av,
                                float* __restrict__ aq, int E) {
  int e = blockIdx.x; if (e >= E) return;
  int j = threadIdx.x; if (j >= 480) return;
  int d = dst[e];
  if (j < 128) {
    int h = j >> 5;
    float w = attn[(size_t)e * 4 + h] / (nodesum[(size_t)d * 4 + h] + 1e-9f);
    atomicAdd(&as_[(size_t)d * 128 + j], ms[(size_t)e * 128 + j] * w);
  } else if (j < 320) {
    int u = j - 128; int m = u / 3; int h = m >> 4;
    float w = attn[(size_t)e * 4 + h] / (nodesum[(size_t)d * 4 + h] + 1e-9f);
    atomicAdd(&av[(size_t)d * 192 + u], mv[(size_t)e * 192 + u] * w);
  } else {
    int u = j - 320; int m = u / 5; int h = m >> 3;
    float w = attn[(size_t)e * 4 + h] / (nodesum[(size_t)d * 4 + h] + 1e-9f);
    atomicAdd(&aq[(size_t)d * 160 + u], mq[(size_t)e * 160 + u] * w);
  }
}

// out[n][k][c] (+)= (sum_m X[n][m][c] * W[m][k]) * (gate ? gate[n][k] : 1)
__global__ void sg_vec_linear(const float* __restrict__ X, const float* __restrict__ W,
                              const float* __restrict__ gate, float* __restrict__ out,
                              int n, int M, int K, int C, int accumulate) {
  size_t i = (size_t)blockIdx.x * blockDim.x + threadIdx.x;
  if (i >= (size_t)n * K * C) return;
  size_t nid = i / ((size_t)K * C); int u = (int)(i % ((size_t)K * C));
  int k = u / C, c = u % C;
  const float* xp = X + nid * (size_t)M * C + c;
  float acc = 0.0f;
  for (int m = 0; m < M; ++m) acc += xp[m * C] * W[m * K + k];
  if (gate) acc *= gate[nid * (size_t)K + k];
  if (accumulate) out[i] += acc; else out[i] = acc;
}

__global__ __launch_bounds__(256)
void sg_ln512(const float* __restrict__ feat, const float* __restrict__ g,
              const float* __restrict__ b, bf16_t* __restrict__ out, int n) {
  __shared__ float red[256];
  int nid = blockIdx.x; if (nid >= n) return;
  int t = threadIdx.x;
  float a0 = feat[(size_t)nid * 512 + t];
  float a1 = feat[(size_t)nid * 512 + 256 + t];
  red[t] = a0 + a1; __syncthreads();
  for (int st = 128; st > 0; st >>= 1) { if (t < st) red[t] += red[t + st]; __syncthreads(); }
  float mean = red[0] / 512.0f; __syncthreads();
  float d0 = a0 - mean, d1 = a1 - mean;
  red[t] = d0 * d0 + d1 * d1; __syncthreads();
  for (int st = 128; st > 0; st >>= 1) { if (t < st) red[t] += red[t + st]; __syncthreads(); }
  float var = red[0] / 512.0f; __syncthreads();
  float r = rsqrtf(var + 1e-5f);
  out[(size_t)nid * 512 + t]       = f2bf(d0 * r * g[t] + b[t]);
  out[(size_t)nid * 512 + 256 + t] = f2bf(d1 * r * g[256 + t] + b[256 + t]);
}

__global__ void sg_head(const bf16_t* __restrict__ h1, const float* __restrict__ Wh2,
                        const float* __restrict__ t, float* __restrict__ out, int n) {
  size_t i = (size_t)blockIdx.x * blockDim.x + threadIdx.x;
  if (i >= (size_t)n * 3) return;
  size_t nid = i / 3; int c = (int)(i % 3);
  float acc = 0.0f;
  for (int k = 0; k < 512; ++k) acc += bf2f(h1[nid * 512 + k]) * Wh2[k * 3 + c];
  float tt = t[nid];
  float lmc = -0.25f * tt * tt * (20.0f - 0.1f) - 0.5f * tt * 0.1f;
  float std_ = sqrtf(1.0f - __expf(2.0f * lmc));
  out[i] = -acc / std_;
}

// ---------------------------------------------------------------------------
extern "C" void kernel_launch(void* const* d_in, const int* in_sizes, int n_in,
                              void* d_out, int out_size, void* d_ws, size_t ws_size,
                              hipStream_t stream) {
  const float* pos   = (const float*)d_in[0];
  const float* tptr  = (const float*)d_in[1];
  const int*   f_in  = (const int*)d_in[2];
  const int*   esrc  = (const int*)d_in[4];
  const int*   edst  = (const int*)d_in[5];
  const float* W_embed = (const float*)d_in[6];
  const float* g1s = (const float*)d_in[10];
  const float* g1v = (const float*)d_in[11];
  const float* g1q = (const float*)d_in[12];
  const float* Wvw = (const float*)d_in[19];
  const float* Wqw = (const float*)d_in[20];
  const float* Af  = (const float*)d_in[21];
  const float* Wov = (const float*)d_in[23];
  const float* Woq = (const float*)d_in[24];
  const float* g2s = (const float*)d_in[25];
  const float* g2v = (const float*)d_in[26];
  const float* g2q = (const float*)d_in[27];
  const float* W1v = (const float*)d_in[31];
  const float* W1q = (const float*)d_in[32];
  const float* W2v = (const float*)d_in[34];
  const float* W2q = (const float*)d_in[35];
  const float* ln_g = (const float*)d_in[37];
  const float* ln_b = (const float*)d_in[38];
  const float* Wh2  = (const float*)d_in[40];

  const int NN = in_sizes[0] / 3;   // 8192 nodes
  const int E  = in_sizes[4];       // edges

  // ---- workspace bump allocator ----
  char* wsp = (char*)d_ws;
  size_t off = 0;
  auto alloc = [&](size_t bytes) -> void* {
    void* p = wsp + off;
    off = (off + bytes + 255) & ~(size_t)255;
    return p;
  };
  auto nb = [](size_t n, int b) { return (unsigned)((n + (size_t)b - 1) / b); };
  auto cvt = [&](const float* src, bf16_t* dst, size_t n) {
    sg_f32_to_bf16<<<nb(n, 256), 256, 0, stream>>>(src, dst, n);
  };
  // convert + transpose weights: src nmat x (K x N) f32 -> dst nmat x (N x K) bf16
  auto cvtT = [&](const float* src, bf16_t* dst, int K, int N, int nmat) {
    sg_cvt_w_t<<<nb((size_t)K * N * nmat, 256), 256, 0, stream>>>(src, dst, K, N, nmat);
  };
  auto fill = [&](float* p, float v, size_t n) {
    sg_fill<<<nb(n, 256), 256, 0, stream>>>(p, v, n);
  };
  auto gemm = [&](const bf16_t* A, const int* Aidx, const bf16_t* BT, void* C,
                  int M, int Nn, int K, int act, int acc, int obf,
                  const float* emul, int eld, int eoff) {
    dim3 g((M + 127) / 128, (Nn + 63) / 64);
    sg_wmma_gemm<<<g, 256, 0, stream>>>(A, Aidx, BT, C, M, Nn, K, act, acc, obf,
                                        emul, eld, eoff);
  };

  // ---- bf16 transposed weights ----
  bf16_t* Wr1b   = (bf16_t*)alloc(128 * 64 * 2);   cvtT((const float*)d_in[7],  Wr1b, 128, 64, 1);
  bf16_t* Wr2b   = (bf16_t*)alloc(64 * 64 * 2);    cvtT((const float*)d_in[8],  Wr2b, 64, 64, 1);
  bf16_t* Wr3b   = (bf16_t*)alloc(64 * 224 * 2);   cvtT((const float*)d_in[9],  Wr3b, 64, 224, 1);
  bf16_t* Wrad1b = (bf16_t*)alloc(6 * 8192 * 2);   cvtT((const float*)d_in[13], Wrad1b, 128, 64, 6);
  bf16_t* Wrad2b = (bf16_t*)alloc(6 * 20480 * 2);  cvtT((const float*)d_in[14], Wrad2b, 64, 320, 6);
  bf16_t* Wsb    = (bf16_t*)alloc(6 * 16384 * 2);  cvtT((const float*)d_in[15], Wsb, 128, 128, 6);
  bf16_t* Wsdb   = (bf16_t*)alloc(6 * 16384 * 2);  cvtT((const float*)d_in[16], Wsdb, 128, 128, 6);
  bf16_t* Wsvb   = (bf16_t*)alloc(6 * 8192 * 2);   cvtT((const float*)d_in[17], Wsvb, 128, 64, 6);
  bf16_t* Wsqb   = (bf16_t*)alloc(6 * 4096 * 2);   cvtT((const float*)d_in[18], Wsqb, 128, 32, 6);
  bf16_t* Wosb   = (bf16_t*)alloc(6 * 16384 * 2);  cvtT((const float*)d_in[22], Wosb, 128, 128, 6);
  bf16_t* W1sb   = (bf16_t*)alloc(6 * 16384 * 2);  cvtT((const float*)d_in[28], W1sb, 128, 128, 6);
  bf16_t* W1gvb  = (bf16_t*)alloc(6 * 8192 * 2);   cvtT((const float*)d_in[29], W1gvb, 128, 64, 6);
  bf16_t* W1gqb  = (bf16_t*)alloc(6 * 4096 * 2);   cvtT((const float*)d_in[30], W1gqb, 128, 32, 6);
  bf16_t* W2sb   = (bf16_t*)alloc(6 * 16384 * 2);  cvtT((const float*)d_in[33], W2sb, 128, 128, 6);
  bf16_t* Wfb    = (bf16_t*)alloc(128 * 512 * 2);  cvtT((const float*)d_in[36], Wfb, 128, 512, 1);
  bf16_t* Wh1b   = (bf16_t*)alloc(512 * 512 * 2);  cvtT((const float*)d_in[39], Wh1b, 512, 512, 1);

  // ---- node/edge state buffers ----
  float*  s    = (float*)alloc((size_t)NN * 128 * 4);
  float*  v    = (float*)alloc((size_t)NN * 192 * 4);
  float*  q    = (float*)alloc((size_t)NN * 160 * 4);
  float*  ys   = (float*)alloc((size_t)NN * 128 * 4);
  bf16_t* ysb  = (bf16_t*)alloc((size_t)NN * 128 * 2);
  float*  yv   = (float*)alloc((size_t)NN * 192 * 4);
  float*  yq   = (float*)alloc((size_t)NN * 160 * 4);
  float*  sh   = (float*)alloc((size_t)E * 8 * 4);
  bf16_t* rbf  = (bf16_t*)alloc((size_t)E * 128 * 2);
  bf16_t* h1e  = (bf16_t*)alloc((size_t)E * 64 * 2);
  bf16_t* h2e  = (bf16_t*)alloc((size_t)E * 64 * 2);
  float*  wr   = (float*)alloc((size_t)E * 320 * 4);
  float*  ms   = (float*)alloc((size_t)E * 128 * 4);
  float*  svb  = (float*)alloc((size_t)E * 64 * 4);
  float*  sqb  = (float*)alloc((size_t)E * 32 * 4);
  float*  mv   = (float*)alloc((size_t)E * 192 * 4);
  float*  mq   = (float*)alloc((size_t)E * 160 * 4);
  float*  logits = (float*)alloc((size_t)E * 4 * 4);
  float*  attn   = (float*)alloc((size_t)E * 4 * 4);
  float*  nodemax = (float*)alloc((size_t)NN * 4 * 4);
  float*  nodesum = (float*)alloc((size_t)NN * 4 * 4);
  float*  as_  = (float*)alloc((size_t)NN * 128 * 4);
  float*  av   = (float*)alloc((size_t)NN * 192 * 4);
  float*  aq   = (float*)alloc((size_t)NN * 160 * 4);
  bf16_t* asb  = (bf16_t*)alloc((size_t)NN * 128 * 2);
  bf16_t* smid = (bf16_t*)alloc((size_t)NN * 128 * 2);
  float*  gvb  = (float*)alloc((size_t)NN * 64 * 4);
  float*  gqb  = (float*)alloc((size_t)NN * 32 * 4);
  float*  vmid = (float*)alloc((size_t)NN * 192 * 4);
  float*  qmid = (float*)alloc((size_t)NN * 160 * 4);
  float*  feat = (float*)alloc((size_t)NN * 512 * 4);
  bf16_t* featb = (bf16_t*)alloc((size_t)NN * 512 * 2);
  bf16_t* h1n   = (bf16_t*)alloc((size_t)NN * 512 * 2);
  (void)ws_size; (void)n_in; (void)out_size;

  // ---- embedding + geometry + initial aggregation ----
  sg_node_embed<<<NN, 128, 0, stream>>>(tptr, f_in, W_embed, s, NN);
  fill(v, 0.0f, (size_t)NN * 192);
  fill(q, 0.0f, (size_t)NN * 160);
  sg_edge_geom<<<E, 128, 0, stream>>>(pos, esrc, edst, sh, rbf, E);
  gemm(rbf, nullptr, Wr1b, h1e, E, 64, 128, ACT_SILU, 0, 1, nullptr, 0, 0);
  gemm(h1e, nullptr, Wr2b, h2e, E, 64, 64, ACT_SILU, 0, 1, nullptr, 0, 0);
  gemm(h2e, nullptr, Wr3b, wr, E, 224, 64, ACT_NONE, 0, 0, nullptr, 0, 0);
  sg_scatter_init<<<E, 512, 0, stream>>>(wr, sh, edst, s, v, q, E);

  // ---- layers ----
  for (int L = 0; L < 6; ++L) {
    sg_eqln<<<NN, 128, 0, stream>>>(s, v, q, g1s + L * 128, g1v + L * 64,
                                    g1q + L * 32, ys, ysb, yv, yq, NN);
    // radial edge weights
    gemm(rbf, nullptr, Wrad1b + (size_t)L * 8192, h1e, E, 64, 128, ACT_SILU, 0, 1, nullptr, 0, 0);
    gemm(h1e, nullptr, Wrad2b + (size_t)L * 20480, wr, E, 320, 64, ACT_NONE, 0, 0, nullptr, 0, 0);
    // m_s = (ys[src]@Ws + ys[dst]@Wsd) * aw0   (gather + scale fused into GEMM)
    gemm(ysb, esrc, Wsb + (size_t)L * 16384, ms, E, 128, 128, ACT_NONE, 0, 0, nullptr, 0, 0);
    gemm(ysb, edst, Wsdb + (size_t)L * 16384, ms, E, 128, 128, ACT_NONE, 1, 0, wr, 320, 0);
    // m_v / m_q scalar parts
    gemm(ysb, esrc, Wsvb + (size_t)L * 8192, svb, E, 64, 128, ACT_NONE, 0, 0, nullptr, 0, 0);
    gemm(ysb, esrc, Wsqb + (size_t)L * 4096, sqb, E, 32, 128, ACT_NONE, 0, 0, nullptr, 0, 0);
    sg_mv_edge<<<nb((size_t)E * 192, 256), 256, 0, stream>>>(svb, wr, sh, yv, esrc, Wvw + (size_t)L * 4096, mv, E);
    sg_mq_edge<<<nb((size_t)E * 160, 256), 256, 0, stream>>>(sqb, wr, sh, yq, esrc, Wqw + (size_t)L * 1024, mq, E);
    // attention (segment softmax over edge_dst)
    sg_logits<<<nb((size_t)E * 4, 256), 256, 0, stream>>>(ms, Af + (size_t)L * 128, logits, E);
    fill(nodemax, -1e30f, (size_t)NN * 4);
    fill(nodesum, 0.0f, (size_t)NN * 4);
    sg_segmax<<<nb((size_t)E * 4, 256), 256, 0, stream>>>(logits, edst, nodemax, E);
    sg_segexp<<<nb((size_t)E * 4, 256), 256, 0, stream>>>(logits, edst, nodemax, nodesum, attn, E);
    fill(as_, 0.0f, (size_t)NN * 128);
    fill(av, 0.0f, (size_t)NN * 192);
    fill(aq, 0.0f, (size_t)NN * 160);
    sg_scatter_attn<<<E, 512, 0, stream>>>(ms, mv, mq, attn, nodesum, edst, as_, av, aq, E);
    // output projections (residual)
    cvt(as_, asb, (size_t)NN * 128);
    gemm(asb, nullptr, Wosb + (size_t)L * 16384, s, NN, 128, 128, ACT_NONE, 1, 0, nullptr, 0, 0);
    sg_vec_linear<<<nb((size_t)NN * 192, 256), 256, 0, stream>>>(av, Wov + (size_t)L * 4096, nullptr, v, NN, 64, 64, 3, 1);
    sg_vec_linear<<<nb((size_t)NN * 160, 256), 256, 0, stream>>>(aq, Woq + (size_t)L * 1024, nullptr, q, NN, 32, 32, 5, 1);
    // gated FFN
    sg_eqln<<<NN, 128, 0, stream>>>(s, v, q, g2s + L * 128, g2v + L * 64,
                                    g2q + L * 32, ys, ysb, yv, yq, NN);
    gemm(ysb, nullptr, W1sb + (size_t)L * 16384, smid, NN, 128, 128, ACT_SILU, 0, 1, nullptr, 0, 0);
    gemm(ysb, nullptr, W1gvb + (size_t)L * 8192, gvb, NN, 64, 128, ACT_SIG, 0, 0, nullptr, 0, 0);
    gemm(ysb, nullptr, W1gqb + (size_t)L * 4096, gqb, NN, 32, 128, ACT_SIG, 0, 0, nullptr, 0, 0);
    sg_vec_linear<<<nb((size_t)NN * 192, 256), 256, 0, stream>>>(yv, W1v + (size_t)L * 4096, gvb, vmid, NN, 64, 64, 3, 0);
    sg_vec_linear<<<nb((size_t)NN * 160, 256), 256, 0, stream>>>(yq, W1q + (size_t)L * 1024, gqb, qmid, NN, 32, 32, 5, 0);
    gemm(smid, nullptr, W2sb + (size_t)L * 16384, s, NN, 128, 128, ACT_NONE, 1, 0, nullptr, 0, 0);
    sg_vec_linear<<<nb((size_t)NN * 192, 256), 256, 0, stream>>>(vmid, W2v + (size_t)L * 4096, nullptr, v, NN, 64, 64, 3, 1);
    sg_vec_linear<<<nb((size_t)NN * 160, 256), 256, 0, stream>>>(qmid, W2q + (size_t)L * 1024, nullptr, q, NN, 32, 32, 5, 1);
  }

  // ---- head ----
  cvt(s, ysb, (size_t)NN * 128);
  gemm(ysb, nullptr, Wfb, feat, NN, 512, 128, ACT_NONE, 0, 0, nullptr, 0, 0);
  sg_ln512<<<NN, 256, 0, stream>>>(feat, ln_g, ln_b, featb, NN);
  gemm(featb, nullptr, Wh1b, h1n, NN, 512, 512, ACT_SILU, 0, 1, nullptr, 0, 0);
  sg_head<<<nb((size_t)NN * 3, 256), 256, 0, stream>>>(h1n, Wh2, tptr, (float*)d_out, NN);
}
